// MoELayer_39436389712280
// MI455X (gfx1250) — matrix operand order repaired and checked by reading, not compiled, not run on hardware
//
#include <hip/hip_runtime.h>
#include <hip/hip_bf16.h>
#include <math.h>

#define T_TOK 8192
#define DIMV  1024
#define HIDV  1024
#define NEXP  7
#define TOPK  2
#define MTILE 32                              // rows per GEMM tile (2 WMMA fragments)
#define CAP   (TOPK * T_TOK + NEXP * MTILE)   // padded assignment-slot capacity
#define NBLK  512                             // N columns staged per block per chunk

typedef __attribute__((ext_vector_type(16))) __bf16 v16bf;
typedef __attribute__((ext_vector_type(8)))  float  v8f;
typedef __attribute__((ext_vector_type(4)))  unsigned int u32x4;
typedef __attribute__((ext_vector_type(8)))  int i32x8;
typedef __attribute__((ext_vector_type(4)))  int i32x4;

union VU { v16bf v; int4 q[2]; };

#if defined(__HIP_DEVICE_COMPILE__) && __has_builtin(__builtin_amdgcn_tensor_load_to_lds) && __has_builtin(__builtin_amdgcn_s_wait_tensorcnt)
#define HAS_TDM 1
#else
#define HAS_TDM 0
#endif

__device__ __forceinline__ float gelu_exact(float x) {
    return 0.5f * x * (1.0f + erff(x * 0.70710678118654752f));
}

// ---------------- TDM: load one [NBLK rows x 32 K] bf16 tile (32KB) into LDS ----------------
// Weights stored [N][K] with K contiguous; tile row = output column, 32 bf16 per row.
__device__ __forceinline__ void tdm_issue_b(const __hip_bfloat16* gsrc, unsigned int lds_off) {
#if HAS_TDM
    unsigned long long ga = (unsigned long long)(uintptr_t)gsrc;
    u32x4 g0;
    g0.x = 1u;                                        // count=1, user mode, no gather
    g0.y = lds_off;                                   // lds_addr [63:32]
    g0.z = (unsigned int)ga;                          // global_addr [95:64]
    g0.w = (unsigned int)((ga >> 32) & 0x01FFFFFFu) | (2u << 30);  // addr[56:32] | type=2
    i32x8 g1;
    g1[0] = (int)(1u << 16);                          // workgroup_mask=0, data_size=1 (2B)
    g1[1] = (int)(32u << 16);                         // tensor_dim0 = 32  (bits 79:48)
    g1[2] = (int)((unsigned)NBLK << 16);              // tensor_dim1 = 512 (bits 111:80)
    g1[3] = (int)(32u << 16);                         // tile_dim0 = 32    (bits 127:112)
    g1[4] = NBLK;                                     // tile_dim1 = 512, tile_dim2 = 0
    g1[5] = DIMV;                                     // tensor_dim0_stride = 1024 (low 32)
    g1[6] = 0;
    g1[7] = 0;
    i32x4 z4 = {0, 0, 0, 0};
#if __clang_major__ >= 23
    i32x8 z8 = {0, 0, 0, 0, 0, 0, 0, 0};
    __builtin_amdgcn_tensor_load_to_lds(g0, g1, z4, z4, z8, 0);
#else
    __builtin_amdgcn_tensor_load_to_lds(g0, g1, z4, z4, 0);
#endif
#else
    (void)gsrc; (void)lds_off;
#endif
}

// ---------------- wave GEMM core: 32 x 64-per-wave (MS=2, NS=4), B staged in LDS ----------
// a0row/a1row: per-lane A row pointers (K contiguous bf16). wtBlock = weights + nBase*Kdim.
template<int KDIM>
__device__ __forceinline__ void gemm_core(const __hip_bfloat16* __restrict__ a0row,
                                          const __hip_bfloat16* __restrict__ a1row,
                                          const __hip_bfloat16* __restrict__ wtBlock,
                                          __hip_bfloat16* ldsB /* [2][NBLK*32] */,
                                          int wave, int lane, v8f acc[2][4]) {
    const int half = lane >> 4;
    const int nchunks = KDIM / 32;
#if HAS_TDM
    if (wave == 0) tdm_issue_b(wtBlock, (unsigned int)(uintptr_t)ldsB);
#endif
    for (int i = 0; i < nchunks; ++i) {
        const int kc = i * 32;
#if HAS_TDM
        if (wave == 0) {
            if (i + 1 < nchunks) {
                tdm_issue_b(wtBlock + (size_t)(i + 1) * 32,
                            (unsigned int)(uintptr_t)(ldsB + ((i + 1) & 1) * (NBLK * 32)));
                __builtin_amdgcn_s_wait_tensorcnt((short)1);
            } else {
                __builtin_amdgcn_s_wait_tensorcnt((short)0);
            }
        }
        const __hip_bfloat16* bbuf = ldsB + (i & 1) * (NBLK * 32);
#else
        // fallback: cooperative copy of the 512x32 tile (2 rows of 64B per thread)
        {
            int tid = (int)threadIdx.x;
            for (int r = tid; r < NBLK; r += 256) {
                const int4* src = (const int4*)(wtBlock + (size_t)r * KDIM + kc);
                int4* dst = (int4*)(ldsB + r * 32);
                dst[0] = src[0];
                dst[1] = src[1];
                dst[2] = src[2];
                dst[3] = src[3];
            }
        }
        const __hip_bfloat16* bbuf = ldsB;
#endif
        __syncthreads();
        VU a0, a1;
        a0.q[0] = *(const int4*)(a0row + kc + half * 8);
        a0.q[1] = *(const int4*)(a0row + kc + 16 + half * 8);
        a1.q[0] = *(const int4*)(a1row + kc + half * 8);
        a1.q[1] = *(const int4*)(a1row + kc + 16 + half * 8);
        const int nl = lane & 15;
#pragma unroll
        for (int j = 0; j < 4; ++j) {
            int nloc = wave * 64 + j * 16 + nl;
            const __hip_bfloat16* bp = bbuf + nloc * 32 + half * 8;
            VU b;
            b.q[0] = *(const int4*)(bp);
            b.q[1] = *(const int4*)(bp + 16);
            acc[0][j] = __builtin_amdgcn_wmma_f32_16x16x32_bf16(
                false, a0.v, false, b.v, (short)0, acc[0][j], false, false);
            acc[1][j] = __builtin_amdgcn_wmma_f32_16x16x32_bf16(
                false, a1.v, false, b.v, (short)0, acc[1][j], false, false);
        }
        __syncthreads();
    }
}

__device__ __forceinline__ void zero_acc(v8f acc[2][4]) {
#pragma unroll
    for (int i = 0; i < 2; ++i)
#pragma unroll
        for (int j = 0; j < 4; ++j)
#pragma unroll
            for (int q = 0; q < 8; ++q) acc[i][j][q] = 0.0f;
}

// ---------------- init: zero meta + padded lists ----------------
__global__ void k_init(int* __restrict__ list_tok, float* __restrict__ list_w,
                       int* __restrict__ meta, int cap) {
    int i = blockIdx.x * blockDim.x + threadIdx.x;
    if (i < cap) { list_tok[i] = 0; list_w[i] = 0.0f; }
    if (i < 32) meta[i] = 0;
}

// ---------------- f32 -> bf16 ----------------
__global__ void k_cvt(const float* __restrict__ src, __hip_bfloat16* __restrict__ dst, int n) {
    int i = blockIdx.x * blockDim.x + threadIdx.x;
    if (i < n) dst[i] = __float2bfloat16(src[i]);
}

// ---------------- transpose+convert: dst[mat][c][r] = src[mat][r][c] ----------------
__global__ void k_tcvt(const float* __restrict__ src, __hip_bfloat16* __restrict__ dst,
                       int R, int C, int total) {
    int i = blockIdx.x * blockDim.x + threadIdx.x;
    if (i >= total) return;
    int rc  = R * C;
    int mat = i / rc, rem = i - mat * rc;
    int r = rem / C, c = rem - r * C;
    dst[(size_t)mat * rc + (size_t)c * R + r] = __float2bfloat16(src[i]);
}

// ---------------- router: one wave per token ----------------
__global__ void k_router(const float* __restrict__ x, const float* __restrict__ gw,
                         const float* __restrict__ ebias,
                         int* __restrict__ tok_e, float* __restrict__ tok_w,
                         int* __restrict__ counts /* = meta[0..7] */) {
    int wave = threadIdx.x >> 5, lane = threadIdx.x & 31;
    int t = blockIdx.x * 8 + wave;
    if (t >= T_TOK) return;
    const float* xr = x + (size_t)t * DIMV;
    float acc[NEXP];
#pragma unroll
    for (int e = 0; e < NEXP; ++e) acc[e] = 0.0f;
    for (int i = lane; i < DIMV; i += 32) {
        float xv = xr[i];
#pragma unroll
        for (int e = 0; e < NEXP; ++e) acc[e] += xv * gw[e * DIMV + i];
    }
#pragma unroll
    for (int e = 0; e < NEXP; ++e)
#pragma unroll
        for (int off = 16; off > 0; off >>= 1)
            acc[e] += __shfl_xor(acc[e], off, 32);
    if (lane == 0) {
        float l[NEXP];
#pragma unroll
        for (int e = 0; e < NEXP; ++e) l[e] = acc[e] + ebias[e];
        int i0 = 0; float v0 = l[0];
#pragma unroll
        for (int e = 1; e < NEXP; ++e) if (l[e] > v0) { v0 = l[e]; i0 = e; }
        int i1 = (i0 == 0) ? 1 : 0; float v1 = l[i1];
#pragma unroll
        for (int e = 0; e < NEXP; ++e) if (e != i0 && l[e] > v1) { v1 = l[e]; i1 = e; }
        float e1 = expf(v1 - v0);
        float inv = 1.0f / (1.0f + e1);
        tok_e[2 * t]     = i0; tok_w[2 * t]     = inv;
        tok_e[2 * t + 1] = i1; tok_w[2 * t + 1] = e1 * inv;
        atomicAdd(&counts[i0], 1);
        atomicAdd(&counts[i1], 1);
    }
}

// meta layout: [0..7] counts, [8..15] offsets (32-aligned), [16..23] cursor, [24..31] ntiles(32-row)
__global__ void k_scan(int* __restrict__ meta) {
    if (threadIdx.x == 0 && blockIdx.x == 0) {
        int off = 0;
        for (int e = 0; e < NEXP; ++e) {
            meta[8 + e] = off;
            int nt = (meta[e] + MTILE - 1) / MTILE;
            meta[24 + e] = nt;
            off += nt * MTILE;
        }
    }
}

__global__ void k_build(const int* __restrict__ tok_e, const float* __restrict__ tok_w,
                        int* __restrict__ meta, int* __restrict__ list_tok,
                        float* __restrict__ list_w) {
    int t = blockIdx.x * blockDim.x + threadIdx.x;
    if (t >= T_TOK) return;
#pragma unroll
    for (int k = 0; k < TOPK; ++k) {
        int e = tok_e[2 * t + k];
        int p = atomicAdd(&meta[16 + e], 1);
        int idx = meta[8 + e] + p;
        list_tok[idx] = t;
        list_w[idx]   = tok_w[2 * t + k];
    }
}

// ---------------- shared expert GEMM1: h = gelu(x @ sw1 + sb1) ----------------
__global__ __launch_bounds__(256) void k_gemm_sh1(const __hip_bfloat16* __restrict__ xb,
                                                  const __hip_bfloat16* __restrict__ w1t,
                                                  const float* __restrict__ b1,
                                                  __hip_bfloat16* __restrict__ h) {
    __shared__ __align__(128) __hip_bfloat16 ldsB[2 * NBLK * 32];
    int lane = threadIdx.x & 31, wave = threadIdx.x >> 5;
    int half = lane >> 4, nl = lane & 15;
    int m0 = blockIdx.x * MTILE;
    int nBase = blockIdx.y * NBLK;
    const __hip_bfloat16* a0 = xb + (size_t)(m0 + nl) * DIMV;
    const __hip_bfloat16* a1 = xb + (size_t)(m0 + 16 + nl) * DIMV;
    v8f acc[2][4];
    zero_acc(acc);
    gemm_core<DIMV>(a0, a1, w1t + (size_t)nBase * DIMV, ldsB, wave, lane, acc);
#pragma unroll
    for (int mi = 0; mi < 2; ++mi)
#pragma unroll
        for (int j = 0; j < 4; ++j) {
            int n = nBase + wave * 64 + j * 16 + nl;
            float bias = b1[n];
#pragma unroll
            for (int v = 0; v < 8; ++v) {
                int m = m0 + mi * 16 + v + half * 8;
                h[(size_t)m * HIDV + n] = __float2bfloat16(gelu_exact(acc[mi][j][v] + bias));
            }
        }
}

// ---------------- shared expert GEMM2: out = h @ sw2 + sb2 (f32 store) ----------------
__global__ __launch_bounds__(256) void k_gemm_sh2(const __hip_bfloat16* __restrict__ h,
                                                  const __hip_bfloat16* __restrict__ w2t,
                                                  const float* __restrict__ b2,
                                                  float* __restrict__ out) {
    __shared__ __align__(128) __hip_bfloat16 ldsB[2 * NBLK * 32];
    int lane = threadIdx.x & 31, wave = threadIdx.x >> 5;
    int half = lane >> 4, nl = lane & 15;
    int m0 = blockIdx.x * MTILE;
    int nBase = blockIdx.y * NBLK;
    const __hip_bfloat16* a0 = h + (size_t)(m0 + nl) * HIDV;
    const __hip_bfloat16* a1 = h + (size_t)(m0 + 16 + nl) * HIDV;
    v8f acc[2][4];
    zero_acc(acc);
    gemm_core<HIDV>(a0, a1, w2t + (size_t)nBase * HIDV, ldsB, wave, lane, acc);
#pragma unroll
    for (int mi = 0; mi < 2; ++mi)
#pragma unroll
        for (int j = 0; j < 4; ++j) {
            int n = nBase + wave * 64 + j * 16 + nl;
            float bias = b2[n];
#pragma unroll
            for (int v = 0; v < 8; ++v) {
                int m = m0 + mi * 16 + v + half * 8;
                out[(size_t)m * DIMV + n] = acc[mi][j][v] + bias;
            }
        }
}

// ---------------- routed GEMM1 over gathered tokens ----------------
__global__ __launch_bounds__(256) void k_gemm_rt1(const __hip_bfloat16* __restrict__ xb,
                                                  const __hip_bfloat16* __restrict__ rw1t,
                                                  const float* __restrict__ rb1,
                                                  const int* __restrict__ meta,
                                                  const int* __restrict__ list_tok,
                                                  __hip_bfloat16* __restrict__ hrt) {
    __shared__ __align__(128) __hip_bfloat16 ldsB[2 * NBLK * 32];
    int e = blockIdx.z;
    if ((int)blockIdx.x >= meta[24 + e]) return;
    int lane = threadIdx.x & 31, wave = threadIdx.x >> 5;
    int half = lane >> 4, nl = lane & 15;
    int base = meta[8 + e] + blockIdx.x * MTILE;
    int nBase = blockIdx.y * NBLK;
    const __hip_bfloat16* a0 = xb + (size_t)list_tok[base + nl] * DIMV;
    const __hip_bfloat16* a1 = xb + (size_t)list_tok[base + 16 + nl] * DIMV;
    const __hip_bfloat16* wt = rw1t + (size_t)e * HIDV * DIMV + (size_t)nBase * DIMV;
    const float* b1 = rb1 + e * HIDV;
    v8f acc[2][4];
    zero_acc(acc);
    gemm_core<DIMV>(a0, a1, wt, ldsB, wave, lane, acc);
#pragma unroll
    for (int mi = 0; mi < 2; ++mi)
#pragma unroll
        for (int j = 0; j < 4; ++j) {
            int n = nBase + wave * 64 + j * 16 + nl;
            float bias = b1[n];
#pragma unroll
            for (int v = 0; v < 8; ++v) {
                int s = base + mi * 16 + v + half * 8;   // assignment slot (C row)
                hrt[(size_t)s * HIDV + n] = __float2bfloat16(gelu_exact(acc[mi][j][v] + bias));
            }
        }
}

// ---------------- routed GEMM2: out += w * (h_r @ rw2 + rb2), scatter-add ----------------
__global__ __launch_bounds__(256) void k_gemm_rt2(const __hip_bfloat16* __restrict__ hrt,
                                                  const __hip_bfloat16* __restrict__ rw2t,
                                                  const float* __restrict__ rb2,
                                                  const int* __restrict__ meta,
                                                  const int* __restrict__ list_tok,
                                                  const float* __restrict__ list_w,
                                                  float* __restrict__ out) {
    __shared__ __align__(128) __hip_bfloat16 ldsB[2 * NBLK * 32];
    int e = blockIdx.z;
    if ((int)blockIdx.x >= meta[24 + e]) return;
    int lane = threadIdx.x & 31, wave = threadIdx.x >> 5;
    int half = lane >> 4, nl = lane & 15;
    int base = meta[8 + e] + blockIdx.x * MTILE;
    int nBase = blockIdx.y * NBLK;
    const __hip_bfloat16* a0 = hrt + (size_t)(base + nl) * HIDV;
    const __hip_bfloat16* a1 = hrt + (size_t)(base + 16 + nl) * HIDV;
    const __hip_bfloat16* wt = rw2t + (size_t)e * DIMV * HIDV + (size_t)nBase * HIDV;
    const float* b2 = rb2 + e * DIMV;
    v8f acc[2][4];
    zero_acc(acc);
    gemm_core<HIDV>(a0, a1, wt, ldsB, wave, lane, acc);
    int   tk[2][8];
    float wg[2][8];
#pragma unroll
    for (int mi = 0; mi < 2; ++mi)
#pragma unroll
        for (int v = 0; v < 8; ++v) {
            int s = base + mi * 16 + v + half * 8;
            tk[mi][v] = list_tok[s];
            wg[mi][v] = list_w[s];     // padded rows carry weight 0 -> contribute nothing
        }
#pragma unroll
    for (int mi = 0; mi < 2; ++mi)
#pragma unroll
        for (int j = 0; j < 4; ++j) {
            int n = nBase + wave * 64 + j * 16 + nl;
            float bias = b2[n];
#pragma unroll
            for (int v = 0; v < 8; ++v) {
                atomicAdd(&out[(size_t)tk[mi][v] * DIMV + n],
                          (acc[mi][j][v] + bias) * wg[mi][v]);
            }
        }
}

extern "C" void kernel_launch(void* const* d_in, const int* in_sizes, int n_in,
                              void* d_out, int out_size, void* d_ws, size_t ws_size,
                              hipStream_t stream) {
    (void)in_sizes; (void)n_in; (void)out_size; (void)ws_size;
    const float* x      = (const float*)d_in[0];
    const float* gate_w = (const float*)d_in[1];
    const float* ebias  = (const float*)d_in[2];
    const float* sw1    = (const float*)d_in[3];
    const float* sb1    = (const float*)d_in[4];
    const float* sw2    = (const float*)d_in[5];
    const float* sb2    = (const float*)d_in[6];
    const float* rw1    = (const float*)d_in[7];
    const float* rb1    = (const float*)d_in[8];
    const float* rw2    = (const float*)d_in[9];
    const float* rb2    = (const float*)d_in[10];
    float* out = (float*)d_out;

    uint8_t* base = (uint8_t*)d_ws;
    size_t o = 0;
    auto alloc = [&](size_t bytes) {
        uint8_t* p = base + o;
        o = (o + bytes + 255) & ~(size_t)255;
        return p;
    };
    __hip_bfloat16* xb   = (__hip_bfloat16*)alloc((size_t)T_TOK * DIMV * 2);
    __hip_bfloat16* sw1t = (__hip_bfloat16*)alloc((size_t)HIDV * DIMV * 2);
    __hip_bfloat16* sw2t = (__hip_bfloat16*)alloc((size_t)DIMV * HIDV * 2);
    __hip_bfloat16* rw1t = (__hip_bfloat16*)alloc((size_t)NEXP * HIDV * DIMV * 2);
    __hip_bfloat16* rw2t = (__hip_bfloat16*)alloc((size_t)NEXP * DIMV * HIDV * 2);
    __hip_bfloat16* hsh  = (__hip_bfloat16*)alloc((size_t)T_TOK * HIDV * 2);
    __hip_bfloat16* hrt  = (__hip_bfloat16*)alloc((size_t)CAP * HIDV * 2);
    int*   tok_e    = (int*)  alloc((size_t)T_TOK * 2 * 4);
    float* tok_w    = (float*)alloc((size_t)T_TOK * 2 * 4);
    int*   list_tok = (int*)  alloc((size_t)CAP * 4);
    float* list_w   = (float*)alloc((size_t)CAP * 4);
    int*   meta     = (int*)  alloc(128);

    k_init<<<(CAP + 255) / 256, 256, 0, stream>>>(list_tok, list_w, meta, CAP);
    k_cvt<<<(T_TOK * DIMV + 255) / 256, 256, 0, stream>>>(x, xb, T_TOK * DIMV);
    k_tcvt<<<(DIMV * HIDV + 255) / 256, 256, 0, stream>>>(sw1, sw1t, DIMV, HIDV, DIMV * HIDV);
    k_tcvt<<<(HIDV * DIMV + 255) / 256, 256, 0, stream>>>(sw2, sw2t, HIDV, DIMV, HIDV * DIMV);
    k_tcvt<<<(NEXP * DIMV * HIDV + 255) / 256, 256, 0, stream>>>(rw1, rw1t, DIMV, HIDV, NEXP * DIMV * HIDV);
    k_tcvt<<<(NEXP * HIDV * DIMV + 255) / 256, 256, 0, stream>>>(rw2, rw2t, HIDV, DIMV, NEXP * HIDV * DIMV);
    k_router<<<T_TOK / 8, 256, 0, stream>>>(x, gate_w, ebias, tok_e, tok_w, meta);
    k_scan<<<1, 32, 0, stream>>>(meta);
    k_build<<<(T_TOK + 255) / 256, 256, 0, stream>>>(tok_e, tok_w, meta, list_tok, list_w);

    dim3 gsh(T_TOK / MTILE, DIMV / NBLK);            // (256, 2)
    k_gemm_sh1<<<gsh, 256, 0, stream>>>(xb, sw1t, sb1, hsh);
    k_gemm_sh2<<<gsh, 256, 0, stream>>>(hsh, sw2t, sb2, out);

    dim3 grt(T_TOK / MTILE, DIMV / NBLK, NEXP);      // (256, 2, 7)
    k_gemm_rt1<<<grt, 256, 0, stream>>>(xb, rw1t, rb1, meta, list_tok, hrt);
    k_gemm_rt2<<<grt, 256, 0, stream>>>(hrt, rw2t, rb2, meta, list_tok, list_w, out);
}
